// Qwen3NextSparseMoeBlock_43963285242744
// MI455X (gfx1250) — matrix operand order repaired and checked by reading, not compiled, not run on hardware
//
#include <hip/hip_runtime.h>
#include <hip/hip_bf16.h>

// ---------------------------------------------------------------------------
// Qwen3-Next sparse MoE block for gfx1250 (MI455X), wave32, WMMA bf16 path.
// Each GEMM wave computes a 16x64 output tile (A-fragment reuse, 4/8 accums).
// ---------------------------------------------------------------------------

typedef __bf16 bf16_t;
typedef bf16_t v16bf __attribute__((ext_vector_type(16)));
typedef bf16_t v8bf  __attribute__((ext_vector_type(8)));
typedef float  v8f   __attribute__((ext_vector_type(8)));

#define T_TOK   2048
#define DIM     1024
#define MOE_HID 512
#define N_EXP   16
#define SH_HID  2048
#define ENT_CAP 4352      // 4096 real (T*K) + up to 16*15 padding, 16-aligned
#define MAX_TILES (ENT_CAP/16)

// ---- WMMA fragment loaders ------------------------------------------------
// A (16x32 bf16, MxK): lane r=lane&15 holds row M=r; lanes 0-15 K=0-15,
//   lanes 16-31 K=16-31 -> one contiguous 32B run at k0 = 16*(lane>>4).
__device__ __forceinline__ v16bf load_frag_a(const bf16_t* __restrict__ p) {
  v8bf lo = *reinterpret_cast<const v8bf*>(p);
  v8bf hi = *reinterpret_cast<const v8bf*>(p + 8);
  return __builtin_shufflevector(lo, hi, 0,1,2,3,4,5,6,7,8,9,10,11,12,13,14,15);
}
// B (32x16 bf16, KxN): lane n=lane&15 holds col N=n; lanes 0-15 K=0-15,
//   lanes 16-31 K=16-31 -> one contiguous 32B run at k0 = 16*(lane>>4).
__device__ __forceinline__ v16bf load_frag_b(const bf16_t* __restrict__ p) {
  v8bf lo = *reinterpret_cast<const v8bf*>(p);
  v8bf hi = *reinterpret_cast<const v8bf*>(p + 8);
  return __builtin_shufflevector(lo, hi, 0,1,2,3,4,5,6,7,8,9,10,11,12,13,14,15);
}

__device__ __forceinline__ v8f wmma_bf16(v16bf a, v16bf b, v8f c) {
  return __builtin_amdgcn_wmma_f32_16x16x32_bf16(false, a, false, b, (short)0, c,
                                                 false, false);
}

__device__ __forceinline__ float wsum(float v) {
  #pragma unroll
  for (int o = 16; o > 0; o >>= 1) v += __shfl_xor(v, o, 32);
  return v;
}

__device__ __forceinline__ float silu_f(float g) {
  return g * (1.f / (1.f + __expf(-g)));
}

// ---- 0) fp32 -> bf16 conversion -------------------------------------------

__global__ void cvt_f32_bf16(const float* __restrict__ s, bf16_t* __restrict__ d, int n) {
  int i = blockIdx.x * blockDim.x + threadIdx.x;
  int stride = gridDim.x * blockDim.x;
  for (; i < n; i += stride) d[i] = (bf16_t)s[i];
}

// ---- 1) router: softmax over 16 experts, top-2 (renormalized), sigmoid gate

__global__ void router_kernel(const float* __restrict__ x,
                              const float* __restrict__ Wg,
                              const float* __restrict__ Wsig,
                              float* __restrict__ scores,
                              int* __restrict__ inds,
                              float* __restrict__ sig) {
  const int lane = threadIdx.x & 31;
  const int wave = threadIdx.x >> 5;
  const int t = blockIdx.x * 4 + wave;          // blockDim = 128 -> 4 tokens/blk
  if (t >= T_TOK) return;

  float xv[32];
  const float* xr = x + (size_t)t * DIM;
  #pragma unroll
  for (int j = 0; j < 32; ++j) xv[j] = xr[lane + 32 * j];

  float logit = -__builtin_inff();
  for (int e = 0; e < N_EXP; ++e) {
    const float* wr = Wg + (size_t)e * DIM;
    float s = 0.f;
    #pragma unroll
    for (int j = 0; j < 32; ++j) s = fmaf(xv[j], wr[lane + 32 * j], s);
    s = wsum(s);
    if (lane == e) logit = s;                   // lane e holds logit[e]
  }
  float sg = 0.f;
  #pragma unroll
  for (int j = 0; j < 32; ++j) sg = fmaf(xv[j], Wsig[lane + 32 * j], sg);
  sg = wsum(sg);

  // softmax over lanes 0..15
  float m = logit;
  #pragma unroll
  for (int o = 16; o > 0; o >>= 1) m = fmaxf(m, __shfl_xor(m, o, 32));
  float ex = (lane < N_EXP) ? __expf(logit - m) : 0.f;
  float den = wsum(ex);
  float p = ex / den;

  // top-1 (value, lowest index on tie)
  float v = p; int idx = lane;
  #pragma unroll
  for (int o = 16; o > 0; o >>= 1) {
    float ov = __shfl_xor(v, o, 32);
    int   oi = __shfl_xor(idx, o, 32);
    if (ov > v || (ov == v && oi < idx)) { v = ov; idx = oi; }
  }
  float v1 = v; int i1 = idx;
  // top-2
  float p2 = (lane == i1) ? -__builtin_inff() : p;
  v = p2; idx = lane;
  #pragma unroll
  for (int o = 16; o > 0; o >>= 1) {
    float ov = __shfl_xor(v, o, 32);
    int   oi = __shfl_xor(idx, o, 32);
    if (ov > v || (ov == v && oi < idx)) { v = ov; idx = oi; }
  }
  float v2 = v; int i2 = idx;

  if (lane == 0) {
    float ss = v1 + v2;
    scores[t * 2 + 0] = v1 / ss;
    scores[t * 2 + 1] = v2 / ss;
    inds[t * 2 + 0] = i1;
    inds[t * 2 + 1] = i2;
    sig[t] = 1.f / (1.f + __expf(-sg));
  }
}

// ---- 2) deterministic bucketization by expert (1 block, 16 waves) ----------

__global__ void bucket_kernel(const int* __restrict__ inds,
                              const float* __restrict__ scores,
                              int* __restrict__ tok_ent,
                              float* __restrict__ scl_ent,
                              int* __restrict__ pos,
                              int* __restrict__ tile_exp) {
  __shared__ int cnt[N_EXP];
  __shared__ int off[N_EXP];
  const int lane = threadIdx.x & 31;
  const int e    = threadIdx.x >> 5;            // blockDim = 512 -> wave == expert

  for (int i = threadIdx.x; i < MAX_TILES; i += 512) tile_exp[i] = -1;

  // count matches for this expert
  int c = 0;
  for (int base = 0; base < T_TOK * 2; base += 32) {
    int f = (inds[base + lane] == e) ? 1 : 0;
    int s = f;
    #pragma unroll
    for (int o = 16; o > 0; o >>= 1) s += __shfl_xor(s, o, 32);
    c += s;
  }
  if (lane == 0) cnt[e] = c;
  __syncthreads();

  if (threadIdx.x == 0) {
    int o0 = 0;
    for (int k = 0; k < N_EXP; ++k) { off[k] = o0; o0 += (cnt[k] + 15) & ~15; }
  }
  __syncthreads();

  const int base_off = off[e];
  int cursor = 0;
  for (int base = 0; base < T_TOK * 2; base += 32) {
    int i = base + lane;
    int f = (inds[i] == e) ? 1 : 0;
    int incl = f;
    #pragma unroll
    for (int o = 1; o < 32; o <<= 1) {
      int n = __shfl_up(incl, o, 32);
      if (lane >= o) incl += n;
    }
    int excl = incl - f;
    int tot  = __shfl(incl, 31, 32);
    if (f) {
      int idx = base_off + cursor + excl;
      tok_ent[idx] = i >> 1;                    // token
      scl_ent[idx] = scores[i];                 // renormalized top-k weight
      pos[i] = idx;                             // (t,k) -> entry
    }
    cursor += tot;
  }
  // pad bucket to multiple of 16 with zero-weight rows
  int padded = (cursor + 15) & ~15;
  for (int j = cursor + lane; j < padded; j += 32) {
    tok_ent[base_off + j] = 0;
    scl_ent[base_off + j] = 0.f;
  }
  for (int tt = (base_off >> 4) + lane; tt < ((base_off + padded) >> 4); tt += 32)
    tile_exp[tt] = e;
}

// ---- 3) expert gate/up GEMM + SwiGLU (gathered rows), 16x64 per wave -------
// grid (MAX_TILES, MOE_HID/256), block 128 (4 waves, 64 cols each)

__global__ void moe_gateup_kernel(const bf16_t* __restrict__ xb,
                                  const bf16_t* __restrict__ Wgt,
                                  const bf16_t* __restrict__ Wup,
                                  const int* __restrict__ tok_ent,
                                  const int* __restrict__ tile_exp,
                                  bf16_t* __restrict__ hbuf) {
  const int lane = threadIdx.x & 31;
  const int wave = threadIdx.x >> 5;
  const int tile = blockIdx.x;
  const int e = tile_exp[tile];
  if (e < 0) return;
  const int half = lane >> 4, r15 = lane & 15;
  const int entry0 = tile * 16;
  const int n0 = (blockIdx.y * 4 + wave) * 64;

  const int row = tok_ent[entry0 + r15];
  const bf16_t* A = xb + (size_t)row * DIM + half * 16;
  const bf16_t* Bg[4]; const bf16_t* Bu[4];
  #pragma unroll
  for (int j = 0; j < 4; ++j) {
    size_t brow = (size_t)e * MOE_HID + (n0 + j * 16 + r15);
    Bg[j] = Wgt + brow * DIM + half * 16;
    Bu[j] = Wup + brow * DIM + half * 16;
  }

  v8f accg[4] = {}; v8f accu[4] = {};
  for (int ks = 0; ks < DIM; ks += 32) {
    v16bf a = load_frag_a(A + ks);
    #pragma unroll
    for (int j = 0; j < 4; ++j) {
      v16bf bg = load_frag_b(Bg[j] + ks);
      accg[j] = wmma_bf16(a, bg, accg[j]);
    }
    #pragma unroll
    for (int j = 0; j < 4; ++j) {
      v16bf bu = load_frag_b(Bu[j] + ks);
      accu[j] = wmma_bf16(a, bu, accu[j]);
    }
  }
  #pragma unroll
  for (int j = 0; j < 4; ++j) {
    #pragma unroll
    for (int rr = 0; rr < 8; ++rr) {
      int mrow = rr + 8 * half;
      float hv = silu_f(accg[j][rr]) * accu[j][rr];
      hbuf[(size_t)(entry0 + mrow) * MOE_HID + n0 + j * 16 + r15] = (bf16_t)hv;
    }
  }
}

// ---- 4) expert down GEMM, score-scaled per-entry output, 16x64 per wave ----
// grid (MAX_TILES, DIM/256), block 128

__global__ void moe_down_kernel(const bf16_t* __restrict__ hbuf,
                                const bf16_t* __restrict__ Wdn,
                                const float* __restrict__ scl_ent,
                                const int* __restrict__ tile_exp,
                                float* __restrict__ yent) {
  const int lane = threadIdx.x & 31;
  const int wave = threadIdx.x >> 5;
  const int tile = blockIdx.x;
  const int e = tile_exp[tile];
  if (e < 0) return;
  const int half = lane >> 4, r15 = lane & 15;
  const int entry0 = tile * 16;
  const int n0 = (blockIdx.y * 4 + wave) * 64;

  const bf16_t* A = hbuf + (size_t)(entry0 + r15) * MOE_HID + half * 16;
  const bf16_t* B[4];
  #pragma unroll
  for (int j = 0; j < 4; ++j)
    B[j] = Wdn + ((size_t)e * DIM + (n0 + j * 16 + r15)) * MOE_HID + half * 16;

  v8f acc[4] = {};
  for (int ks = 0; ks < MOE_HID; ks += 32) {
    v16bf a = load_frag_a(A + ks);
    #pragma unroll
    for (int j = 0; j < 4; ++j) {
      v16bf b = load_frag_b(B[j] + ks);
      acc[j] = wmma_bf16(a, b, acc[j]);
    }
  }
  #pragma unroll
  for (int rr = 0; rr < 8; ++rr) {
    int mrow = rr + 8 * half;
    float sc = scl_ent[entry0 + mrow];
    #pragma unroll
    for (int j = 0; j < 4; ++j)
      yent[(size_t)(entry0 + mrow) * DIM + n0 + j * 16 + r15] = acc[j][rr] * sc;
  }
}

// ---- 5) shared expert gate/up + SwiGLU, 16x64 per wave ---------------------
// grid (T/16, SH_HID/256), block 128

__global__ void shared_gateup_kernel(const bf16_t* __restrict__ xb,
                                     const bf16_t* __restrict__ Wgt,
                                     const bf16_t* __restrict__ Wup,
                                     bf16_t* __restrict__ H1) {
  const int lane = threadIdx.x & 31;
  const int wave = threadIdx.x >> 5;
  const int half = lane >> 4, r15 = lane & 15;
  const int row0 = blockIdx.x * 16;
  const int n0 = (blockIdx.y * 4 + wave) * 64;

  const bf16_t* A = xb + (size_t)(row0 + r15) * DIM + half * 16;
  const bf16_t* Bg[4]; const bf16_t* Bu[4];
  #pragma unroll
  for (int j = 0; j < 4; ++j) {
    Bg[j] = Wgt + (size_t)(n0 + j * 16 + r15) * DIM + half * 16;
    Bu[j] = Wup + (size_t)(n0 + j * 16 + r15) * DIM + half * 16;
  }

  v8f accg[4] = {}; v8f accu[4] = {};
  for (int ks = 0; ks < DIM; ks += 32) {
    v16bf a = load_frag_a(A + ks);
    #pragma unroll
    for (int j = 0; j < 4; ++j) {
      v16bf bg = load_frag_b(Bg[j] + ks);
      accg[j] = wmma_bf16(a, bg, accg[j]);
    }
    #pragma unroll
    for (int j = 0; j < 4; ++j) {
      v16bf bu = load_frag_b(Bu[j] + ks);
      accu[j] = wmma_bf16(a, bu, accu[j]);
    }
  }
  #pragma unroll
  for (int j = 0; j < 4; ++j) {
    #pragma unroll
    for (int rr = 0; rr < 8; ++rr) {
      int mrow = rr + 8 * half;
      float hv = silu_f(accg[j][rr]) * accu[j][rr];
      H1[(size_t)(row0 + mrow) * SH_HID + n0 + j * 16 + r15] = (bf16_t)hv;
    }
  }
}

// ---- 6) shared expert down GEMM, 16x64 per wave ----------------------------
// grid (T/16, DIM/256), block 128

__global__ void shared_down_kernel(const bf16_t* __restrict__ H1,
                                   const bf16_t* __restrict__ Wdn,
                                   float* __restrict__ shout) {
  const int lane = threadIdx.x & 31;
  const int wave = threadIdx.x >> 5;
  const int half = lane >> 4, r15 = lane & 15;
  const int row0 = blockIdx.x * 16;
  const int n0 = (blockIdx.y * 4 + wave) * 64;

  const bf16_t* A = H1 + (size_t)(row0 + r15) * SH_HID + half * 16;
  const bf16_t* B[4];
  #pragma unroll
  for (int j = 0; j < 4; ++j)
    B[j] = Wdn + (size_t)(n0 + j * 16 + r15) * SH_HID + half * 16;

  v8f acc[4] = {};
  for (int ks = 0; ks < SH_HID; ks += 32) {
    v16bf a = load_frag_a(A + ks);
    #pragma unroll
    for (int j = 0; j < 4; ++j) {
      v16bf b = load_frag_b(B[j] + ks);
      acc[j] = wmma_bf16(a, b, acc[j]);
    }
  }
  #pragma unroll
  for (int rr = 0; rr < 8; ++rr) {
    int mrow = rr + 8 * half;
    #pragma unroll
    for (int j = 0; j < 4; ++j)
      shout[(size_t)(row0 + mrow) * DIM + n0 + j * 16 + r15] = acc[j][rr];
  }
}

// ---- 7) final combine ------------------------------------------------------

__global__ void combine_kernel(const float* __restrict__ yent,
                               const float* __restrict__ shout,
                               const int* __restrict__ pos,
                               const float* __restrict__ sig,
                               float* __restrict__ out) {
  const int t = blockIdx.x;
  const int p0 = pos[t * 2 + 0];
  const int p1 = pos[t * 2 + 1];
  const float sg = sig[t];
  for (int d = threadIdx.x; d < DIM; d += blockDim.x) {
    out[(size_t)t * DIM + d] = yent[(size_t)p0 * DIM + d]
                             + yent[(size_t)p1 * DIM + d]
                             + sg * shout[(size_t)t * DIM + d];
  }
}

// ---------------------------------------------------------------------------

extern "C" void kernel_launch(void* const* d_in, const int* in_sizes, int n_in,
                              void* d_out, int out_size, void* d_ws, size_t ws_size,
                              hipStream_t stream) {
  (void)in_sizes; (void)n_in; (void)out_size; (void)ws_size;
  const float* x    = (const float*)d_in[0];   // [2,1024,1024]
  const float* Wg   = (const float*)d_in[1];   // [16,1024] router
  const float* Weg  = (const float*)d_in[2];   // [16,512,1024]
  const float* Weu  = (const float*)d_in[3];   // [16,512,1024]
  const float* Wed  = (const float*)d_in[4];   // [16,1024,512]
  const float* Wsgt = (const float*)d_in[5];   // [2048,1024]
  const float* Wsup = (const float*)d_in[6];   // [2048,1024]
  const float* Wsdn = (const float*)d_in[7];   // [1024,2048]
  const float* Wsig = (const float*)d_in[8];   // [1,1024] sigmoid gate
  float* out = (float*)d_out;

  char* w = (char*)d_ws;
  size_t off = 0;
  auto alloc = [&](size_t bytes) -> void* {
    void* p = w + off;
    off = (off + bytes + 255) & ~(size_t)255;
    return p;
  };

  bf16_t* xb    = (bf16_t*)alloc((size_t)T_TOK * DIM * 2);
  bf16_t* wegb  = (bf16_t*)alloc((size_t)N_EXP * MOE_HID * DIM * 2);
  bf16_t* weub  = (bf16_t*)alloc((size_t)N_EXP * MOE_HID * DIM * 2);
  bf16_t* wedb  = (bf16_t*)alloc((size_t)N_EXP * DIM * MOE_HID * 2);
  bf16_t* wsgb  = (bf16_t*)alloc((size_t)SH_HID * DIM * 2);
  bf16_t* wsub  = (bf16_t*)alloc((size_t)SH_HID * DIM * 2);
  bf16_t* wsdb  = (bf16_t*)alloc((size_t)DIM * SH_HID * 2);
  bf16_t* hbuf  = (bf16_t*)alloc((size_t)ENT_CAP * MOE_HID * 2);
  bf16_t* h1    = (bf16_t*)alloc((size_t)T_TOK * SH_HID * 2);
  float*  yent  = (float*)alloc((size_t)ENT_CAP * DIM * 4);
  float*  shout = (float*)alloc((size_t)T_TOK * DIM * 4);
  float*  scores= (float*)alloc((size_t)T_TOK * 2 * 4);
  int*    inds  = (int*)  alloc((size_t)T_TOK * 2 * 4);
  int*    pos   = (int*)  alloc((size_t)T_TOK * 2 * 4);
  float*  sig   = (float*)alloc((size_t)T_TOK * 4);
  int*    tok   = (int*)  alloc((size_t)ENT_CAP * 4);
  float*  scl   = (float*)alloc((size_t)ENT_CAP * 4);
  int*    texp  = (int*)  alloc((size_t)MAX_TILES * 4);

  auto cvt = [&](const float* s, bf16_t* d, int n) {
    int blocks = (n + 255) / 256; if (blocks > 4096) blocks = 4096;
    cvt_f32_bf16<<<blocks, 256, 0, stream>>>(s, d, n);
  };
  cvt(x,    xb,   T_TOK * DIM);
  cvt(Weg,  wegb, N_EXP * MOE_HID * DIM);
  cvt(Weu,  weub, N_EXP * MOE_HID * DIM);
  cvt(Wed,  wedb, N_EXP * DIM * MOE_HID);
  cvt(Wsgt, wsgb, SH_HID * DIM);
  cvt(Wsup, wsub, SH_HID * DIM);
  cvt(Wsdn, wsdb, DIM * SH_HID);

  router_kernel<<<T_TOK / 4, 128, 0, stream>>>(x, Wg, Wsig, scores, inds, sig);
  bucket_kernel<<<1, 512, 0, stream>>>(inds, scores, tok, scl, pos, texp);

  moe_gateup_kernel<<<dim3(MAX_TILES, MOE_HID / 256), 128, 0, stream>>>(
      xb, wegb, weub, tok, texp, hbuf);
  moe_down_kernel<<<dim3(MAX_TILES, DIM / 256), 128, 0, stream>>>(
      hbuf, wedb, scl, texp, yent);

  shared_gateup_kernel<<<dim3(T_TOK / 16, SH_HID / 256), 128, 0, stream>>>(
      xb, wsgb, wsub, h1);
  shared_down_kernel<<<dim3(T_TOK / 16, DIM / 256), 128, 0, stream>>>(
      h1, wsdb, shout);

  combine_kernel<<<T_TOK, 256, 0, stream>>>(yent, shout, pos, sig, out);
}
